// SwinTransformerBlock_34651796144669
// MI455X (gfx1250) — matrix operand order repaired and checked by reading, not compiled, not run on hardware
//
#include <hip/hip_runtime.h>
#include <hip/hip_bf16.h>

typedef _Float16 h16;
typedef __attribute__((ext_vector_type(8)))  _Float16 v8h;
typedef __attribute__((ext_vector_type(16))) _Float16 v16h;
typedef __attribute__((ext_vector_type(8)))  float    v8f;

#define NHEAD 3
#define DIMC  96
#define HDIM  32
#define NTOK  49
#define NPAD  64
#define IMGHW 224
#define LTOT  (224*224)
#define HIDD  384
#define QSCALE 0.17677669529663687f   /* 32^-0.5 */
#define LNEPS 1e-4f

// ---------------- CDNA5 async global->LDS staging helpers ------------------------
// global_load_async_to_lds_b128: per-lane 16B transfer, ASYNCcnt-tracked (ISA §10.7).
// LDS address operand = wave-relative LDS byte offset = low 32 bits of flat pointer.
__device__ __forceinline__ void async_g2l_b128(void* lds_ptr, const void* gptr) {
  const unsigned int       loff  = (unsigned int)(uintptr_t)lds_ptr;
  const unsigned long long gaddr = (unsigned long long)(uintptr_t)gptr;
  asm volatile("global_load_async_to_lds_b128 %0, %1, off"
               :: "v"(loff), "v"(gaddr) : "memory");
}
__device__ __forceinline__ void async_wait0() {
  asm volatile("s_wait_asynccnt 0x0" ::: "memory");
}

// ---------------- WMMA fragment helpers (wave32; contiguous k order, same in A and B) ----

// A 16x32 f16: lane row = lane&15; k = klo..klo+15 contiguous, klo = lane<16?0:16
__device__ __forceinline__ v16h load_a_tile(const h16* p, int ld, int lane) {
  const int row = lane & 15;
  const int klo = (lane & 16) ? 16 : 0;
  const h16* r = p + row * ld + klo;
  v8h lo = *(const v8h*)(r);
  v8h hi = *(const v8h*)(r + 8);
  v16h a;
#pragma unroll
  for (int i = 0; i < 8; ++i) { a[i] = lo[i]; a[i + 8] = hi[i]; }
  return a;
}

// B 32x16 f16: lane col = lane&15; k = klo..klo+15 contiguous, klo = lane<16?0:16
// B[k][n] = src[n*ld + k]  (i.e. src holds B^T row-major: out = A * src^T)
__device__ __forceinline__ v16h load_b_tile(const h16* p, int ld, int lane) {
  const int n   = lane & 15;
  const int klo = (lane & 16) ? 16 : 0;
  const h16* r = p + n * ld + klo;
  v8h lo = *(const v8h*)(r);
  v8h hi = *(const v8h*)(r + 8);
  v16h b;
#pragma unroll
  for (int i = 0; i < 8; ++i) { b[i] = lo[i]; b[i + 8] = hi[i]; }
  return b;
}

// B 32x16 where memory holds B row-major (stride ld between k rows): strided gather
__device__ __forceinline__ v16h load_b_strided(const h16* p, int ld, int lane) {
  const int n   = lane & 15;
  const int klo = (lane & 16) ? 16 : 0;
  v16h b;
#pragma unroll
  for (int k = 0; k < 16; ++k) b[k] = p[(klo + k) * ld + n];
  return b;
}

__device__ __forceinline__ v8f wmma16(v16h a, v16h b, v8f c) {
  return __builtin_amdgcn_wmma_f32_16x16x32_f16(false, a, false, b, (short)0, c,
                                                false, false);
}

// ---------------- K0: f32 -> f16 weight convert ---------------------------------
__global__ void k_f32_to_f16(const float* __restrict__ src, h16* __restrict__ dst, int n) {
  int i = blockIdx.x * 256 + threadIdx.x;
  if (i < n) dst[i] = (h16)src[i];
}

// ---------------- K1: LN1 + cyclic shift + window partition ----------------------
// One wave per padded window row (8192 windows * 64 rows). 4 waves / block.
__global__ __launch_bounds__(128) void k_ln1_part(
    const float* __restrict__ x, const float* __restrict__ nm,
    const float* __restrict__ g, const float* __restrict__ b,
    h16* __restrict__ xw, float* __restrict__ nmw) {
  const int wave = threadIdx.x >> 5, lane = threadIdx.x & 31;
  const int row = blockIdx.x * 4 + wave;      // 0 .. 8192*64-1
  const int win = row >> 6, n = row & 63;
  h16* dst = xw + (size_t)row * DIMC;
  if (n >= NTOK) {                            // zero-pad rows 49..63
    for (int i = lane; i < DIMC; i += 32) dst[i] = (h16)0.f;
    if (lane == 0) nmw[row] = 0.f;
    return;
  }
  const int bb = win >> 10, wi = win & 1023, wh = wi >> 5, wc = wi & 31;
  int hp = wh * 7 + n / 7 + 3; if (hp >= IMGHW) hp -= IMGHW;   // roll(-SHIFT) gather
  int wp = wc * 7 + n % 7 + 3; if (wp >= IMGHW) wp -= IMGHW;
  const size_t l = (size_t)bb * LTOT + (size_t)hp * IMGHW + wp;
  const float* px = x + l * DIMC;
  float v0 = px[lane], v1 = px[lane + 32], v2 = px[lane + 64];
  float s = v0 + v1 + v2;
#pragma unroll
  for (int o = 16; o; o >>= 1) s += __shfl_xor(s, o, 32);
  const float mean = s * (1.f / 96.f);
  const float d0 = v0 - mean, d1 = v1 - mean, d2 = v2 - mean;
  float q = d0 * d0 + d1 * d1 + d2 * d2;
#pragma unroll
  for (int o = 16; o; o >>= 1) q += __shfl_xor(q, o, 32);
  const float rstd = rsqrtf(q * (1.f / 96.f) + LNEPS);
  dst[lane]      = (h16)(d0 * rstd * g[lane]      + b[lane]);
  dst[lane + 32] = (h16)(d1 * rstd * g[lane + 32] + b[lane + 32]);
  dst[lane + 64] = (h16)(d2 * rstd * g[lane + 64] + b[lane + 64]);
  if (lane == 0) nmw[row] = nm[l];
}

// ---------------- K2: fused window attention (QKV + attn + proj + residual) ------
__global__ __launch_bounds__(128) void k_attn(
    const h16* __restrict__ xw, const float* __restrict__ nmw,
    const h16* __restrict__ wq, const float* __restrict__ bq,
    const float* __restrict__ rpbt,
    const h16* __restrict__ wpj, const float* __restrict__ bpj,
    const float* __restrict__ x, float* __restrict__ x2) {
  __shared__ h16   qs[NPAD * DIMC];          // q (scaled); reused per-head as attn-out
  __shared__ h16   ks[NPAD * DIMC];
  __shared__ h16   vs[NPAD * DIMC];
  __shared__ float attn[NPAD * NPAD];        // stage B: A-tile staging; then logits/probs
  __shared__ float rpbl[169 * NHEAD];
  __shared__ float nml[NPAD];
  __shared__ float rinv[NPAD];

  const int tid = threadIdx.x, lane = tid & 31, wave = tid >> 5;
  const int win = blockIdx.x;
  const int bb = win >> 10, wi = win & 1023, wh = wi >> 5, wc = wi & 31;

  for (int i = tid; i < 169 * NHEAD; i += 128) rpbl[i] = rpbt[i];
  for (int i = tid; i < NPAD; i += 128) nml[i] = nmw[win * NPAD + i];
  if (tid == 0) { __builtin_prefetch(wq, 0, 3); __builtin_prefetch(wpj, 0, 3); }

  // ---- async-stage the 64x96 f16 window tile into LDS (aliases attn buffer) ----
  h16* xs = (h16*)attn;                              // 12 KB <= 16 KB region, dead until stage C
  {
    const char* gsrc = (const char*)(xw + (size_t)win * NPAD * DIMC);
    char* lbase = (char*)xs;
    for (int c = tid; c < 768; c += 128)             // 768 x 16B = 12288 B
      async_g2l_b128(lbase + c * 16, gsrc + c * 16);
    async_wait0();
  }
  __syncthreads();

  // ---- QKV: (64x96) @ Wqkv^T (288x96) ; cols 0..95=q, 96..191=k, 192..287=v ----
  for (int nt = wave; nt < 18; nt += 4) {
    v8f acc[4] = {};
    for (int k = 0; k < 3; ++k) {
      v16h bt = load_b_tile(wq + nt * 16 * DIMC + k * 32, DIMC, lane);
#pragma unroll
      for (int mt = 0; mt < 4; ++mt) {
        v16h at = load_a_tile(xs + mt * 16 * DIMC + k * 32, DIMC, lane);
        acc[mt] = wmma16(at, bt, acc[mt]);
      }
    }
    const int cg = nt * 16 + (lane & 15);
    const int s = cg / DIMC, cw = cg % DIMC;
    const float bias = bq[cg];
    h16* dstbuf = (s == 0) ? qs : (s == 1) ? ks : vs;
    const float scl = (s == 0) ? QSCALE : 1.f;
#pragma unroll
    for (int mt = 0; mt < 4; ++mt)
#pragma unroll
      for (int r = 0; r < 8; ++r) {
        const int m = mt * 16 + r + ((lane & 16) ? 8 : 0);
        dstbuf[m * DIMC + cw] = (h16)((acc[mt][r] + bias) * scl);
      }
  }
  __syncthreads();

  for (int h = 0; h < NHEAD; ++h) {
    // ---- logits = q_h (64x32) @ k_h^T -> 64x64 f32 ----
    for (int t = wave; t < 16; t += 4) {
      const int mt = t >> 2, nt = t & 3;
      v16h a  = load_a_tile(qs + (mt * 16) * DIMC + h * HDIM, DIMC, lane);
      v16h bt = load_b_tile(ks + (nt * 16) * DIMC + h * HDIM, DIMC, lane);
      v8f acc = {};
      acc = wmma16(a, bt, acc);
      const int col = nt * 16 + (lane & 15);
#pragma unroll
      for (int r = 0; r < 8; ++r)
        attn[(mt * 16 + r + ((lane & 16) ? 8 : 0)) * NPAD + col] = acc[r];
    }
    __syncthreads();

    // ---- softmax row-wise with rpb + nuclei + shift-mask (labels analytic) ----
    if (tid < NPAD) {
      const int n = tid;
      float* rowf = attn + n * NPAD;
      h16*   rowh = (h16*)rowf;
      if (n < NTOK) {
        const int nh_ = n / 7, nw_ = n % 7;
        const int hpn = wh * 7 + nh_, wpn = wc * 7 + nw_;
        const int labn = (hpn < 217 ? 0 : hpn < 221 ? 1 : 2) * 3 +
                         (wpn < 217 ? 0 : wpn < 221 ? 1 : 2);
        const float nmn = nml[n];
        float mx = -3.0e38f;
        for (int m = 0; m < NTOK; ++m) {
          const int mh = m / 7, mw = m % 7;
          const int hpm = wh * 7 + mh, wpm = wc * 7 + mw;
          const int labm = (hpm < 217 ? 0 : hpm < 221 ? 1 : 2) * 3 +
                           (wpm < 217 ? 0 : wpm < 221 ? 1 : 2);
          const int idx = (nh_ - mh + 6) * 13 + (nw_ - mw + 6);
          float v = rowf[m] + rpbl[idx * NHEAD + h] + nmn + nml[m] +
                    ((labn == labm) ? 0.f : -100.f);
          rowf[m] = v;
          mx = fmaxf(mx, v);
        }
        float sum = 0.f;
        for (int m = 0; m < NPAD; ++m) {          // in-place f32 -> f16 (safe: m/2 <= m)
          const float pv = (m < NTOK) ? __expf(rowf[m] - mx) : 0.f;
          sum += pv;
          rowh[m] = (h16)pv;
        }
        rinv[n] = 1.f / sum;                      // normalization folded into P@V epilogue
      } else {
        for (int m = 0; m < NPAD; ++m) rowh[m] = (h16)0.f;
        rinv[n] = 1.f;
      }
    }
    __syncthreads();

    // ---- out_h = P (64x64 f16) @ v_h (64x32) ; write into qs cols [h*32, h*32+32) ----
    const h16* Ph = (const h16*)attn;             // row stride 128 halves
    for (int t = wave; t < 8; t += 4) {
      const int mt = t >> 1, nt = t & 1;
      v8f acc = {};
#pragma unroll
      for (int k = 0; k < 2; ++k) {
        v16h a  = load_a_tile(Ph + (mt * 16) * (2 * NPAD) + k * 32, 2 * NPAD, lane);
        v16h bt = load_b_strided(vs + (k * 32) * DIMC + h * HDIM + nt * 16, DIMC, lane);
        acc = wmma16(a, bt, acc);
      }
      const int d = nt * 16 + (lane & 15);
#pragma unroll
      for (int r = 0; r < 8; ++r) {
        const int m = mt * 16 + r + ((lane & 16) ? 8 : 0);
        qs[m * DIMC + h * HDIM + d] = (h16)(acc[r] * rinv[m]);
      }
    }
    __syncthreads();
  }

  // ---- proj (96x96) + bias + residual, scatter through inverse shift/partition ----
  for (int t = wave; t < 24; t += 4) {
    const int mt = t / 6, nt = t % 6;
    v8f acc = {};
    for (int k = 0; k < 3; ++k) {
      v16h a  = load_a_tile(qs + (mt * 16) * DIMC + k * 32, DIMC, lane);
      v16h bt = load_b_tile(wpj + nt * 16 * DIMC + k * 32, DIMC, lane);
      acc = wmma16(a, bt, acc);
    }
    const int col = nt * 16 + (lane & 15);
    const float bias = bpj[col];
#pragma unroll
    for (int r = 0; r < 8; ++r) {
      const int m = mt * 16 + r + ((lane & 16) ? 8 : 0);
      if (m < NTOK) {
        int hp = wh * 7 + m / 7 + 3; if (hp >= IMGHW) hp -= IMGHW;
        int wp = wc * 7 + m % 7 + 3; if (wp >= IMGHW) wp -= IMGHW;
        const size_t off = ((size_t)bb * LTOT + (size_t)hp * IMGHW + wp) * DIMC + col;
        x2[off] = acc[r] + bias + x[off];
      }
    }
  }
}

// ---------------- K3: LN2(x2) -> f16 ---------------------------------------------
__global__ __launch_bounds__(128) void k_ln2(
    const float* __restrict__ x2, const float* __restrict__ g,
    const float* __restrict__ b, h16* __restrict__ h2) {
  const int wave = threadIdx.x >> 5, lane = threadIdx.x & 31;
  const size_t row = (size_t)blockIdx.x * 4 + wave;
  const float* px = x2 + row * DIMC;
  float v0 = px[lane], v1 = px[lane + 32], v2 = px[lane + 64];
  float s = v0 + v1 + v2;
#pragma unroll
  for (int o = 16; o; o >>= 1) s += __shfl_xor(s, o, 32);
  const float mean = s * (1.f / 96.f);
  const float d0 = v0 - mean, d1 = v1 - mean, d2 = v2 - mean;
  float q = d0 * d0 + d1 * d1 + d2 * d2;
#pragma unroll
  for (int o = 16; o; o >>= 1) q += __shfl_xor(q, o, 32);
  const float rstd = rsqrtf(q * (1.f / 96.f) + LNEPS);
  h16* dst = h2 + row * DIMC;
  dst[lane]      = (h16)(d0 * rstd * g[lane]      + b[lane]);
  dst[lane + 32] = (h16)(d1 * rstd * g[lane + 32] + b[lane + 32]);
  dst[lane + 64] = (h16)(d2 * rstd * g[lane + 64] + b[lane + 64]);
}

// ---------------- K4: fused MLP  fc1 + GELU(exact) + fc2 + residual --------------
// One wave per 16-row group; A-tile async-staged in LDS, fc1 activations in LDS.
__global__ __launch_bounds__(128) void k_mlp(
    const h16* __restrict__ h2, const h16* __restrict__ w1, const float* __restrict__ b1,
    const h16* __restrict__ w2, const float* __restrict__ b2,
    const float* __restrict__ x2, float* __restrict__ out) {
  __shared__ h16 gbuf[4][16 * HIDD];
  __shared__ h16 abuf[4][16 * DIMC];
  const int lane = threadIdx.x & 31, wave = threadIdx.x >> 5;
  const size_t rowbase = ((size_t)blockIdx.x * 4 + wave) * 16;
  const h16* A = h2 + rowbase * DIMC;
  h16* G  = gbuf[wave];
  h16* Ax = abuf[wave];
  if (threadIdx.x == 0) { __builtin_prefetch(w1, 0, 3); __builtin_prefetch(w2, 0, 3); }

  // async-stage the 16x96 f16 activation tile (3072 B = 192 x 16B, 6 per lane)
  for (int c = lane; c < 192; c += 32)
    async_g2l_b128((char*)Ax + c * 16, (const char*)A + c * 16);
  async_wait0();

  for (int nt = 0; nt < 24; ++nt) {
    v8f acc = {};
    for (int k = 0; k < 3; ++k) {
      v16h a  = load_a_tile(Ax + k * 32, DIMC, lane);
      v16h bt = load_b_tile(w1 + nt * 16 * DIMC + k * 32, DIMC, lane);
      acc = wmma16(a, bt, acc);
    }
    const int col = nt * 16 + (lane & 15);
    const float bias = b1[col];
#pragma unroll
    for (int r = 0; r < 8; ++r) {
      const int m = r + ((lane & 16) ? 8 : 0);
      const float v = acc[r] + bias;
      G[m * HIDD + col] = (h16)(0.5f * v * (1.f + erff(v * 0.70710678118f)));
    }
  }
  __syncthreads();
  for (int nt = 0; nt < 6; ++nt) {
    v8f acc = {};
    for (int k = 0; k < 12; ++k) {
      v16h a  = load_a_tile(G + k * 32, HIDD, lane);
      v16h bt = load_b_tile(w2 + nt * 16 * HIDD + k * 32, HIDD, lane);
      acc = wmma16(a, bt, acc);
    }
    const int col = nt * 16 + (lane & 15);
    const float bias = b2[col];
#pragma unroll
    for (int r = 0; r < 8; ++r) {
      const int m = r + ((lane & 16) ? 8 : 0);
      const size_t off = (rowbase + m) * DIMC + col;
      out[off] = acc[r] + bias + x2[off];
    }
  }
}

// ---------------- launcher --------------------------------------------------------
extern "C" void kernel_launch(void* const* d_in, const int* in_sizes, int n_in,
                              void* d_out, int out_size, void* d_ws, size_t ws_size,
                              hipStream_t stream) {
  (void)in_sizes; (void)n_in; (void)out_size; (void)ws_size;
  const float* x    = (const float*)d_in[0];
  const float* nm   = (const float*)d_in[1];
  const float* ln1g = (const float*)d_in[2];
  const float* ln1b = (const float*)d_in[3];
  const float* qkvw = (const float*)d_in[4];
  const float* qkvb = (const float*)d_in[5];
  const float* rpbt = (const float*)d_in[6];
  const float* projw= (const float*)d_in[7];
  const float* projb= (const float*)d_in[8];
  const float* ln2g = (const float*)d_in[9];
  const float* ln2b = (const float*)d_in[10];
  const float* fc1w = (const float*)d_in[11];
  const float* fc1b = (const float*)d_in[12];
  const float* fc2w = (const float*)d_in[13];
  const float* fc2b = (const float*)d_in[14];
  float* out = (float*)d_out;

  char* p = (char*)d_ws;
  auto alloc = [&](size_t bytes) {
    char* q = p; p += (bytes + 255) & ~(size_t)255; return (void*)q;
  };
  h16*   xw  = (h16*)  alloc((size_t)8192 * 64 * 96 * 2);   // windowed LN1(x), f16
  float* nmw = (float*)alloc((size_t)8192 * 64 * 4);        // windowed nuclei mask
  float* x2  = (float*)alloc((size_t)38535168 * 4);         // x + attn branch
  h16*   h2b = (h16*)  alloc((size_t)38535168 * 2);         // LN2(x2), f16
  h16*   wq  = (h16*)  alloc((size_t)288 * 96 * 2);
  h16*   wp  = (h16*)  alloc((size_t)96 * 96 * 2);
  h16*   w1  = (h16*)  alloc((size_t)384 * 96 * 2);
  h16*   w2  = (h16*)  alloc((size_t)96 * 384 * 2);

  k_f32_to_f16<<<(27648 + 255) / 256, 256, 0, stream>>>(qkvw, wq, 27648);
  k_f32_to_f16<<<(9216  + 255) / 256, 256, 0, stream>>>(projw, wp, 9216);
  k_f32_to_f16<<<(36864 + 255) / 256, 256, 0, stream>>>(fc1w, w1, 36864);
  k_f32_to_f16<<<(36864 + 255) / 256, 256, 0, stream>>>(fc2w, w2, 36864);

  k_ln1_part<<<131072, 128, 0, stream>>>(x, nm, ln1g, ln1b, xw, nmw);
  k_attn<<<8192, 128, 0, stream>>>(xw, nmw, wq, qkvb, rpbt, wp, projb, x, x2);
  k_ln2<<<100352, 128, 0, stream>>>(x2, ln2g, ln2b, h2b);
  k_mlp<<<6272, 128, 0, stream>>>(h2b, w1, fc1b, w2, fc2b, x2, out);
}
